// GCN_17617955848194
// MI455X (gfx1250) — matrix operand order repaired and checked by reading, not compiled, not run on hardware
//
#include <hip/hip_runtime.h>
#include <hip/hip_bf16.h>
#include <math.h>

// ---------------------------------------------------------------------------
// GCN (3-layer, PyG GCNConv + BN + ReLU + log_softmax) for MI455X / gfx1250.
// Dense GEMMs run on wave32 WMMA (v_wmma_f32_16x16x32_f16, f32 accumulate)
// with pre-transposed f16 weights so every B fragment is one contiguous 32B
// load. Edge scatter uses global_atomic_add_f32 on an AGG buffer that is
// pre-initialized with the self-loop + bias term (saves a zero pass and an
// RMW pass per layer). HBM-bound: ~3.5 GB -> ~150us at 23.3 TB/s.
// ---------------------------------------------------------------------------

typedef _Float16 v16h __attribute__((ext_vector_type(16)));
typedef _Float16 v8h  __attribute__((ext_vector_type(8)));
typedef float    v8f  __attribute__((ext_vector_type(8)));

#define BN_EPS 1e-5f

// -------------------------------- utility ----------------------------------

__global__ void k_zero_f32(float* __restrict__ p, long n) {
  long i = (long)blockIdx.x * blockDim.x + threadIdx.x;
  if (i < n) p[i] = 0.0f;
}

__global__ void k_cvt_f32_f16(const float* __restrict__ s, _Float16* __restrict__ d, long n) {
  long i = (long)blockIdx.x * blockDim.x + threadIdx.x;
  if (i < n) d[i] = (_Float16)s[i];
}

// W [K x ncin] f32 (row-major) -> WT [ncout x K] f16 (i.e. W transposed,
// zero-padded to ncout columns). WT[col*K + k] = W[k*ncin + col].
__global__ void k_w_transpose_f16(const float* __restrict__ W, _Float16* __restrict__ WT,
                                  int K, int ncin, int ncout) {
  int i = blockIdx.x * blockDim.x + threadIdx.x;
  int tot = ncout * K;
  if (i >= tot) return;
  int col = i / K, k = i - col * K;
  WT[i] = (col < ncin) ? (_Float16)W[k * ncin + col] : (_Float16)0.0f;
}

// ------------------------------ degree / dinv ------------------------------

__global__ void k_degree(const int* __restrict__ dst, float* __restrict__ deg, int E) {
  int e = blockIdx.x * blockDim.x + threadIdx.x;
  if (e < E) atomicAdd(&deg[dst[e]], 1.0f);
}

__global__ void k_dinv(const float* __restrict__ deg, float* __restrict__ dinv, int n) {
  int i = blockIdx.x * blockDim.x + threadIdx.x;
  if (i < n) dinv[i] = rsqrtf(deg[i] + 1.0f);   // +1 for self-loop
}

// ------------------------------- WMMA GEMM ---------------------------------
// C[nrows x LDW] = A[nrows x 128] (f16) @ B[128 x LDW] (f16, given as WT
// [LDW x 128]), f32 accumulate. 256 threads = 8 waves/block; each wave owns a
// 16-row strip and walks NT 16-column tiles; K=128 as 4 x WMMA_16x16x32_F16.
// Fragment layouts used here (wave32; contiguous K, same in A and B), hi = lane>>4, l16 = lane&15:
//   A (16x32 f16): lane holds M=l16; halves 0..7 = K kk*32+hi*16+0..7,
//                  halves 8..15 = K kk*32+hi*16+8..15
//   B (32x16 f16): lane holds column N=l16; half e -> K = kk*32+hi*16+e
//                  -> contiguous 32B in WT[col][*]
//   C/D (16x16 f32): vgpr r, lane -> M = r + hi*8, N = l16
template <int LDW, int NT>
__global__ __launch_bounds__(256)
void k_gemm_wmma(const _Float16* __restrict__ A, const _Float16* __restrict__ WT,
                 float* __restrict__ C, int nrows) {
  const int lane = threadIdx.x & 31;
  const int wid  = threadIdx.x >> 5;
  const int l16  = lane & 15;
  const int hi   = lane >> 4;

  const int r0 = blockIdx.x * 128 + wid * 16;
  const bool fullStrip = (r0 + 16 <= nrows);    // wave-uniform

  // Clamp A-row for the ragged tail; tail stores are predicated per-row.
  int rowA = r0 + l16;
  if (rowA >= nrows) rowA = nrows - 1;
  const _Float16* __restrict__ arow = A + (size_t)rowA * 128;

  v16h a[4];
#pragma unroll
  for (int kk = 0; kk < 4; ++kk) {
    const v8h lo = *(const v8h*)(arow + kk * 32 + hi * 16);       // K = base+0..7
    const v8h hv = *(const v8h*)(arow + kk * 32 + hi * 16 + 8);   // K = base+8..15
    a[kk] = __builtin_shufflevector(lo, hv, 0, 1, 2, 3, 4, 5, 6, 7,
                                            8, 9, 10, 11, 12, 13, 14, 15);
  }

#pragma unroll
  for (int nt = 0; nt < NT; ++nt) {
    const int col = nt * 16 + l16;
    const _Float16* __restrict__ wcol = WT + (size_t)col * 128;
    v8f c = {};
#pragma unroll
    for (int kk = 0; kk < 4; ++kk) {
      const v16h b = *(const v16h*)(wcol + kk * 32 + hi * 16);    // contiguous 32B
      c = __builtin_amdgcn_wmma_f32_16x16x32_f16(
              /*neg_a=*/false, a[kk], /*neg_b=*/false, b,
              /*c_mod=*/(short)0, c, /*reuse_a=*/false, /*reuse_b=*/false);
    }
    float* __restrict__ crow = C + (size_t)(r0 + hi * 8) * LDW + col;
    if (fullStrip) {
      // common case: 8 unconditional dword stores off one base (imm offsets)
#pragma unroll
      for (int r = 0; r < 8; ++r) crow[(size_t)r * LDW] = c[r];
    } else {
#pragma unroll
      for (int r = 0; r < 8; ++r)
        if (r0 + hi * 8 + r < nrows) crow[(size_t)r * LDW] = c[r];
    }
  }
}

// ---------------------- self-loop + bias init of AGG -----------------------
// AGG[i,f] = H[i,f]*dinv[i]^2 + bias[f]  (bias only for f < nb).
// Runs BEFORE the edge atomics; replaces zero-fill + separate RMW pass.
__global__ void k_init_self_bias(const float* __restrict__ H, float* __restrict__ AGG,
                                 const float* __restrict__ dinv, const float* __restrict__ bias,
                                 int nrows, int F, int nb) {
  long i = (long)blockIdx.x * blockDim.x + threadIdx.x;
  long tot = (long)nrows * F;
  if (i >= tot) return;
  int row = (int)(i / F);
  int f = (int)(i - (long)row * F);
  float di = dinv[row];
  float b = (f < nb) ? bias[f] : 0.0f;
  AGG[i] = H[i] * di * di + b;
}

// ----------------------------- edge aggregation ----------------------------
// AGG[dst] += H[src] * dinv[src]*dinv[dst]; one thread = (edge, 4 features).
__global__ void k_aggregate(const float* __restrict__ H, float* __restrict__ AGG,
                            const int* __restrict__ src, const int* __restrict__ dst,
                            const float* __restrict__ dinv, int E, int F) {
  const int per = F >> 2;
  long gid = (long)blockIdx.x * blockDim.x + threadIdx.x;
  int e = (int)(gid / per);
  if (e >= E) return;
  int f = (int)(gid - (long)e * per) * 4;
  const int s = src[e], d = dst[e];
  const float nrm = dinv[s] * dinv[d];
  const float4 h = *(const float4*)(H + (size_t)s * F + f);
  float* out = AGG + (size_t)d * F + f;
  atomicAdd(out + 0, h.x * nrm);
  atomicAdd(out + 1, h.y * nrm);
  atomicAdd(out + 2, h.z * nrm);
  atomicAdd(out + 3, h.w * nrm);
}

// ------------------------------- batch norm --------------------------------

__global__ void k_bn_stats(const float* __restrict__ X, float* __restrict__ sums,
                           int nrows, int F, int rowsPerBlock) {
  const int f = threadIdx.x;            // blockDim.x == F
  const int r0 = blockIdx.x * rowsPerBlock;
  int r1 = r0 + rowsPerBlock; if (r1 > nrows) r1 = nrows;
  float s = 0.0f, s2 = 0.0f;
  for (int r = r0; r < r1; ++r) {
    const float v = X[(size_t)r * F + f];
    s += v; s2 += v * v;
  }
  atomicAdd(&sums[f], s);
  atomicAdd(&sums[F + f], s2);
}

// y = relu((x-mean)*rsqrt(var+eps)*g + be) -> f16 (feeds next WMMA GEMM)
__global__ void k_bn_apply_relu_f16(const float* __restrict__ X, const float* __restrict__ sums,
                                    const float* __restrict__ g, const float* __restrict__ be,
                                    _Float16* __restrict__ Y, int nrows, int F) {
  long i = (long)blockIdx.x * blockDim.x + threadIdx.x;
  long tot = (long)nrows * F;
  if (i >= tot) return;
  int f = (int)(i % F);
  const float invN = 1.0f / (float)nrows;
  const float mean = sums[f] * invN;
  const float var = sums[F + f] * invN - mean * mean;     // biased, as torch BN
  float y = (X[i] - mean) * rsqrtf(var + BN_EPS) * g[f] + be[f];
  y = fmaxf(y, 0.0f);
  Y[i] = (_Float16)y;
}

// ------------------------------- log-softmax -------------------------------

__global__ void k_log_softmax(const float* __restrict__ X, float* __restrict__ out,
                              int nrows, int ldx, int ncols) {
  int row = blockIdx.x * blockDim.x + threadIdx.x;
  if (row >= nrows) return;
  const float* p = X + (size_t)row * ldx;
  float m = p[0];
  for (int j = 1; j < ncols; ++j) m = fmaxf(m, p[j]);
  float s = 0.0f;
  for (int j = 0; j < ncols; ++j) s += __expf(p[j] - m);
  const float lz = m + __logf(s);
  float* o = out + (size_t)row * ncols;
  for (int j = 0; j < ncols; ++j) o[j] = p[j] - lz;
}

// ------------------------------- launcher ----------------------------------

extern "C" void kernel_launch(void* const* d_in, const int* in_sizes, int n_in,
                              void* d_out, int out_size, void* d_ws, size_t ws_size,
                              hipStream_t stream) {
  (void)n_in; (void)out_size; (void)ws_size;

  const float* x   = (const float*)d_in[0];
  const int*   ei  = (const int*)d_in[1];      // [2,E] -> src = ei[0:E), dst = ei[E:2E)
  const float* W1  = (const float*)d_in[2];
  const float* b1  = (const float*)d_in[3];
  const float* g1  = (const float*)d_in[4];
  const float* be1 = (const float*)d_in[5];
  const float* W2  = (const float*)d_in[6];
  const float* b2  = (const float*)d_in[7];
  const float* g2  = (const float*)d_in[8];
  const float* be2 = (const float*)d_in[9];
  const float* Wf  = (const float*)d_in[10];
  const float* bf  = (const float*)d_in[11];
  float* out = (float*)d_out;

  const int N = in_sizes[0] / 128;
  const int E = in_sizes[1] / 2;
  const int F = 128;
  const int FO = 48;      // final layer padded width (D_OUT=40 -> 48)
  const int NB_OUT = 40;

  const int* src = ei;
  const int* dst = ei + E;

  // --- workspace carve-out (256B aligned) ---
  uintptr_t p = (uintptr_t)d_ws;
  auto carve = [&](size_t bytes) -> void* {
    void* r = (void*)p;
    p += (bytes + 255) & ~(size_t)255;
    return r;
  };
  float*     deg  = (float*)carve((size_t)N * 4);
  float*     dinv = (float*)carve((size_t)N * 4);
  float*     sums = (float*)carve(256 * 4);
  _Float16*  WT   = (_Float16*)carve((size_t)128 * 128 * 2);   // transposed weights
  _Float16*  Xh   = (_Float16*)carve((size_t)N * F * 2);
  float*     H    = (float*)carve((size_t)N * F * 4);
  float*     AGG  = (float*)carve((size_t)N * F * 4);

  const int T = 256;
  auto blocks = [&](long n) { return (int)((n + T - 1) / T); };

  // --- degree / dinv (shared by all layers) ---
  k_zero_f32<<<blocks((long)N), T, 0, stream>>>(deg, N);
  k_degree<<<blocks((long)E), T, 0, stream>>>(dst, deg, E);
  k_dinv<<<blocks((long)N), T, 0, stream>>>(deg, dinv, N);

  // --- input -> f16 ---
  k_cvt_f32_f16<<<blocks((long)N * F), T, 0, stream>>>(x, Xh, (long)N * F);

  const int gemmBlocks = (N + 127) / 128;
  const int bnRows = 256, bnBlocks = (N + bnRows - 1) / bnRows;

  // ============================ layer 1 ============================
  k_w_transpose_f16<<<blocks((long)F * F), T, 0, stream>>>(W1, WT, F, F, F);
  k_gemm_wmma<128, 8><<<gemmBlocks, 256, 0, stream>>>(Xh, WT, H, N);
  k_init_self_bias<<<blocks((long)N * F), T, 0, stream>>>(H, AGG, dinv, b1, N, F, F);
  k_aggregate<<<blocks((long)E * (F / 4)), T, 0, stream>>>(H, AGG, src, dst, dinv, E, F);
  k_zero_f32<<<1, 256, 0, stream>>>(sums, 256);
  k_bn_stats<<<bnBlocks, F, 0, stream>>>(AGG, sums, N, F, bnRows);
  k_bn_apply_relu_f16<<<blocks((long)N * F), T, 0, stream>>>(AGG, sums, g1, be1, Xh, N, F);

  // ============================ layer 2 ============================
  k_w_transpose_f16<<<blocks((long)F * F), T, 0, stream>>>(W2, WT, F, F, F);
  k_gemm_wmma<128, 8><<<gemmBlocks, 256, 0, stream>>>(Xh, WT, H, N);
  k_init_self_bias<<<blocks((long)N * F), T, 0, stream>>>(H, AGG, dinv, b2, N, F, F);
  k_aggregate<<<blocks((long)E * (F / 4)), T, 0, stream>>>(H, AGG, src, dst, dinv, E, F);
  k_zero_f32<<<1, 256, 0, stream>>>(sums, 256);
  k_bn_stats<<<bnBlocks, F, 0, stream>>>(AGG, sums, N, F, bnRows);
  k_bn_apply_relu_f16<<<blocks((long)N * F), T, 0, stream>>>(AGG, sums, g2, be2, Xh, N, F);

  // ============================ layer 3 ============================
  k_w_transpose_f16<<<blocks((long)F * FO), T, 0, stream>>>(Wf, WT, F, NB_OUT, FO);
  k_gemm_wmma<48, 3><<<gemmBlocks, 256, 0, stream>>>(Xh, WT, H, N);
  k_init_self_bias<<<blocks((long)N * FO), T, 0, stream>>>(H, AGG, dinv, bf, N, FO, NB_OUT);
  k_aggregate<<<blocks((long)E * (FO / 4)), T, 0, stream>>>(H, AGG, src, dst, dinv, E, FO);

  // --- log_softmax over first 40 cols -> d_out [N,40] ---
  k_log_softmax<<<blocks((long)N), T, 0, stream>>>(AGG, out, N, FO, NB_OUT);
}
